// DecoderBlock_20023137534920
// MI455X (gfx1250) — compile-verified, matrix operand order checked by reading, not run on hardware
//
#include <hip/hip_runtime.h>
#include <hip/hip_bf16.h>

typedef __bf16 bf16;
typedef __attribute__((ext_vector_type(16))) __bf16 v16bf;
typedef __attribute__((ext_vector_type(8)))  float  v8f;

#define Tn   2048
#define Cn   1024
#define NHn  16
#define HDn  64
#define En   8
#define HIDn 4096

#if defined(__gfx1250__) && __has_builtin(__builtin_amdgcn_tensor_load_to_lds)
#define USE_TDM 1
#else
#define USE_TDM 0
#endif

#if USE_TDM
typedef unsigned int tdm_u32x4 __attribute__((ext_vector_type(4)));
typedef int          tdm_i32x8 __attribute__((ext_vector_type(8)));
typedef int          tdm_i32x4 __attribute__((ext_vector_type(4)));

// Issue a TDM 2-D tile load: tile_dim0=32 (K slab), tile_dim1=rows, elem=2B,
// row stride = Kd elements.  One issue per calling wave; helper waits on
// s_wait_tensorcnt itself; caller adds a workgroup barrier.
__device__ __forceinline__ void tdm_load_tile(const bf16* gsrc, bf16* lds_dst,
                                              int Kd, int rows) {
  unsigned long long ga = (unsigned long long)(size_t)gsrc;
  tdm_u32x4 g0;
  g0[0] = 1u;                                   // count=1, user mode
  g0[1] = (unsigned)(size_t)lds_dst;            // LDS byte offset (low 32b of generic)
  g0[2] = (unsigned)ga;                         // global_addr[31:0]
  g0[3] = (unsigned)((ga >> 32) & 0x01FFFFFFu)  // global_addr[56:32]
        | (2u << 30);                           // type=2 ("image")
  tdm_i32x8 g1;
  g1[0] = (1 << 16);                            // data_size=1 -> 2 bytes
  g1[1] = (int)(((unsigned)Kd & 0xFFFFu) << 16);            // tensor_dim0 lo16
  g1[2] = (int)((((unsigned)Kd >> 16) & 0xFFFFu)            // tensor_dim0 hi16
        | (((unsigned)rows & 0xFFFFu) << 16));              // tensor_dim1 lo16
  g1[3] = (int)(32u << 16);                     // tensor_dim1 hi=0 | tile_dim0=32
  g1[4] = rows & 0xFFFF;                        // tile_dim1=rows, tile_dim2=0
  g1[5] = Kd;                                   // tensor_dim0_stride lo32
  g1[6] = 0;                                    // stride hi | dim1_stride lo
  g1[7] = 0;
  tdm_i32x4 z4 = {};
  tdm_i32x8 z8 = {};
  __builtin_amdgcn_tensor_load_to_lds(g0, g1, z4, z4, z8, 0);
  __builtin_amdgcn_s_wait_tensorcnt((short)0);
}
#endif

union Frag { uint4 q[2]; v16bf v; };

// Load one A/B fragment for v_wmma_f32_16x16x32_bf16 from LDS.
// LDS layout: [major][stride] bf16 (major = m-row for A, n-col for B^T storage).
// Lane = half*16 + lm holds elements k = koff + half*8 + {0..7} and
// k = koff + 16 + half*8 + {0..7} of row `major`.
__device__ __forceinline__ v16bf ld_frag(const bf16* base, int major, int stride,
                                         int koff, int half) {
  const bf16* p = base + (size_t)major * stride + koff + half * 8;
  Frag f;
  f.q[0] = *(const uint4*)(p);
  f.q[1] = *(const uint4*)(p + 16);
  return f.v;
}

__device__ __forceinline__ v8f wmma_bf16(v16bf a, v16bf b, v8f c) {
  return __builtin_amdgcn_wmma_f32_16x16x32_bf16(false, a, false, b, (short)0, c,
                                                 false, false);
}

// ---------------------------------------------------------------------------
// RMSNorm: fp32 in -> bf16 out (GEMM operand form)
// ---------------------------------------------------------------------------
__global__ void rmsnorm_bf16(const float* __restrict__ X, const float* __restrict__ Wg,
                             bf16* __restrict__ Y) {
  const int t = blockIdx.x;
  const int tid = threadIdx.x;
  __shared__ float red[256];
  const float* xr = X + (size_t)t * Cn;
  float s = 0.f;
  for (int c = tid; c < Cn; c += 256) { const float v = xr[c]; s += v * v; }
  red[tid] = s;
  __syncthreads();
  for (int st = 128; st > 0; st >>= 1) {
    if (tid < st) red[tid] += red[tid + st];
    __syncthreads();
  }
  const float inv = rsqrtf(red[0] * (1.f / Cn) + 1e-6f);
  for (int c = tid; c < Cn; c += 256)
    Y[(size_t)t * Cn + c] = (bf16)(xr[c] * inv * Wg[c]);
}

// ---------------------------------------------------------------------------
// Generic bf16 GEMM: D[M,N](f32) = A[M,K](bf16) * B[K,N](f32 -> bf16 in LDS)
// Block = 256 thr (8 waves), tile 128x64, K-step 32.  Wave w: rows w*16..+15,
// all 64 cols (4 accumulators).  A tile via TDM when available.
// ---------------------------------------------------------------------------
__global__ void gemm_bf16_f32(const bf16* __restrict__ A, const float* __restrict__ B,
                              float* __restrict__ D, const float* __restrict__ res,
                              int N, int Kd) {
  __shared__ __align__(16) bf16 sA[128 * 32];
  __shared__ __align__(16) bf16 sB[64 * 32];  // stored [n][k]
  const int tid = threadIdx.x;
  const int lane = tid & 31, w = tid >> 5;
  const int lm = lane & 15, half = lane >> 4;
  const int m0 = blockIdx.y * 128, n0 = blockIdx.x * 64;
  const int bk = tid >> 3, bn = (tid & 7) * 8;
  const int ar = tid >> 1, ac = (tid & 1) * 16;
  v8f zero = {};
  v8f acc[4] = {zero, zero, zero, zero};
  for (int kc = 0; kc < Kd; kc += 32) {
#if !USE_TDM
    uint4 av0 = *(const uint4*)(A + (size_t)(m0 + ar) * Kd + kc + ac);
    uint4 av1 = *(const uint4*)(A + (size_t)(m0 + ar) * Kd + kc + ac + 8);
#endif
    const float* bp = B + (size_t)(kc + bk) * N + n0 + bn;
    float4 b0 = *(const float4*)bp;
    float4 b1 = *(const float4*)(bp + 4);
    if (kc + 32 < Kd) __builtin_prefetch(bp + (size_t)32 * N, 0, 1);
    __syncthreads();
#if USE_TDM
    if (tid < 32) tdm_load_tile(A + (size_t)m0 * Kd + kc, sA, Kd, 128);
#else
    *(uint4*)(sA + ar * 32 + ac) = av0;
    *(uint4*)(sA + ar * 32 + ac + 8) = av1;
#endif
    bf16* q = sB + bk;
    q[(bn + 0) * 32] = (bf16)b0.x; q[(bn + 1) * 32] = (bf16)b0.y;
    q[(bn + 2) * 32] = (bf16)b0.z; q[(bn + 3) * 32] = (bf16)b0.w;
    q[(bn + 4) * 32] = (bf16)b1.x; q[(bn + 5) * 32] = (bf16)b1.y;
    q[(bn + 6) * 32] = (bf16)b1.z; q[(bn + 7) * 32] = (bf16)b1.w;
    __syncthreads();
    v16bf a = ld_frag(sA, w * 16 + lm, 32, 0, half);
#pragma unroll
    for (int j = 0; j < 4; ++j)
      acc[j] = wmma_bf16(a, ld_frag(sB, j * 16 + lm, 32, 0, half), acc[j]);
  }
  const int c0 = n0 + lm;
#pragma unroll
  for (int j = 0; j < 4; ++j) {
#pragma unroll
    for (int v = 0; v < 8; ++v) {
      const int row = m0 + w * 16 + v + 8 * half;
      float xv = acc[j][v];
      if (res) xv += res[(size_t)row * N + c0 + j * 16];
      D[(size_t)row * N + c0 + j * 16] = xv;
    }
  }
}

// ---------------------------------------------------------------------------
// RoPE + repack q,k,v (fp32 [T, NH*HD]) -> bf16 [NH][T][HD] head-major
// ---------------------------------------------------------------------------
__global__ void rope_pack(const float* __restrict__ Q, const float* __restrict__ Kx,
                          const float* __restrict__ V, bf16* __restrict__ Qh,
                          bf16* __restrict__ Kh, bf16* __restrict__ Vh) {
  const int t = blockIdx.x;
  const int tid = threadIdx.x;  // 512
  const int hh = tid >> 5;      // head
  const int d = tid & 31;       // rotary pair index
  float s, c;
  const float inv_freq = __powf(10000.f, -(float)(2 * d) / 64.f);
  __sincosf((float)t * inv_freq, &s, &c);
  const size_t src = (size_t)t * Cn + hh * 64;
  const size_t dst = ((size_t)hh * Tn + t) * 64;
  {
    const float e0 = Q[src + 2 * d], o0 = Q[src + 2 * d + 1];
    Qh[dst + d] = (bf16)(e0 * c - o0 * s);
    Qh[dst + 32 + d] = (bf16)(e0 * s + o0 * c);
  }
  {
    const float e0 = Kx[src + 2 * d], o0 = Kx[src + 2 * d + 1];
    Kh[dst + d] = (bf16)(e0 * c - o0 * s);
    Kh[dst + 32 + d] = (bf16)(e0 * s + o0 * c);
  }
  for (int i = tid; i < Cn; i += 512)
    Vh[((size_t)(i >> 6) * Tn + t) * 64 + (i & 63)] = (bf16)V[(size_t)t * Cn + i];
}

// ---------------------------------------------------------------------------
// Flash attention: block = 128 thr (4 waves), 64 query rows / block,
// 32-key chunks, online softmax, all matmuls via WMMA. Out: bf16 [T, NH*HD].
// ---------------------------------------------------------------------------
__global__ void flash_attn(const bf16* __restrict__ Qh, const bf16* __restrict__ Kh,
                           const bf16* __restrict__ Vh, bf16* __restrict__ Att) {
  const int hh = blockIdx.y;
  const int t0 = blockIdx.x * 64;
  const int tid = threadIdx.x;
  const int lane = tid & 31, w = tid >> 5;
  const int lm = lane & 15, half = lane >> 4;
  __shared__ __align__(16) bf16 sK[32 * 64];      // [key][d]
  __shared__ __align__(16) bf16 sVt[64 * 32];     // [d][key]  (B^T storage)
  __shared__ __align__(16) bf16 sP[4][16 * 32];   // per-wave P tile
  const size_t headoff = (size_t)hh * Tn * 64;
  v16bf qa0, qa1;
  {
    const bf16* qr = Qh + headoff + (size_t)(t0 + w * 16 + lm) * 64;
    Frag f;
    f.q[0] = *(const uint4*)(qr + half * 8);
    f.q[1] = *(const uint4*)(qr + 16 + half * 8);
    qa0 = f.v;
    f.q[0] = *(const uint4*)(qr + 32 + half * 8);
    f.q[1] = *(const uint4*)(qr + 48 + half * 8);
    qa1 = f.v;
  }
  v8f o0 = {}, o1 = {}, o2 = {}, o3 = {};
  float mrow[8], lrow[8];
#pragma unroll
  for (int v = 0; v < 8; ++v) { mrow[v] = -1e30f; lrow[v] = 0.f; }
  const int ldr = tid >> 2;        // 0..31 (key row)
  const int ldc = (tid & 3) * 16;  // 0/16/32/48 (d)
  for (int kc = 0; kc < t0 + 64; kc += 32) {
    __syncthreads();
    {
      const bf16* kp = Kh + headoff + (size_t)(kc + ldr) * 64 + ldc;
      uint4 k0 = *(const uint4*)kp, k1 = *(const uint4*)(kp + 8);
      *(uint4*)(sK + ldr * 64 + ldc) = k0;
      *(uint4*)(sK + ldr * 64 + ldc + 8) = k1;
      const bf16* vp = Vh + headoff + (size_t)(kc + ldr) * 64 + ldc;
#pragma unroll
      for (int i = 0; i < 16; ++i) sVt[(ldc + i) * 32 + ldr] = vp[i];
    }
    __syncthreads();
    v8f s0 = {}, s1 = {};
    s0 = wmma_bf16(qa0, ld_frag(sK, lm, 64, 0, half), s0);
    s0 = wmma_bf16(qa1, ld_frag(sK, lm, 64, 32, half), s0);
    s1 = wmma_bf16(qa0, ld_frag(sK, 16 + lm, 64, 0, half), s1);
    s1 = wmma_bf16(qa1, ld_frag(sK, 16 + lm, 64, 32, half), s1);
    const int key0 = kc + lm, key1 = kc + 16 + lm;
#pragma unroll
    for (int v = 0; v < 8; ++v) {
      const int qrow = t0 + w * 16 + v + 8 * half;
      float a = s0[v] * 0.125f; if (key0 > qrow) a = -1e30f;
      float b = s1[v] * 0.125f; if (key1 > qrow) b = -1e30f;
      float mx = fmaxf(a, b);
      mx = fmaxf(mx, __shfl_xor(mx, 1, 32));
      mx = fmaxf(mx, __shfl_xor(mx, 2, 32));
      mx = fmaxf(mx, __shfl_xor(mx, 4, 32));
      mx = fmaxf(mx, __shfl_xor(mx, 8, 32));
      const float mnew = fmaxf(mrow[v], mx);
      const float scale = __expf(mrow[v] - mnew);
      const float p0 = __expf(a - mnew);
      const float p1 = __expf(b - mnew);
      float ps = p0 + p1;
      ps += __shfl_xor(ps, 1, 32);
      ps += __shfl_xor(ps, 2, 32);
      ps += __shfl_xor(ps, 4, 32);
      ps += __shfl_xor(ps, 8, 32);
      lrow[v] = lrow[v] * scale + ps;
      mrow[v] = mnew;
      o0[v] *= scale; o1[v] *= scale; o2[v] *= scale; o3[v] *= scale;
      const int r = v + 8 * half;
      sP[w][r * 32 + lm] = (bf16)p0;
      sP[w][r * 32 + 16 + lm] = (bf16)p1;
    }
    asm volatile("" ::: "memory");  // keep wave-local P stores before P loads
    v16bf pa = ld_frag(sP[w], lm, 32, 0, half);
    o0 = wmma_bf16(pa, ld_frag(sVt, lm, 32, 0, half), o0);
    o1 = wmma_bf16(pa, ld_frag(sVt, 16 + lm, 32, 0, half), o1);
    o2 = wmma_bf16(pa, ld_frag(sVt, 32 + lm, 32, 0, half), o2);
    o3 = wmma_bf16(pa, ld_frag(sVt, 48 + lm, 32, 0, half), o3);
  }
#pragma unroll
  for (int v = 0; v < 8; ++v) {
    const int trow = t0 + w * 16 + v + 8 * half;
    const float inv = 1.f / lrow[v];
    bf16* orow = Att + (size_t)trow * Cn + hh * 64;
    orow[lm]      = (bf16)(o0[v] * inv);
    orow[16 + lm] = (bf16)(o1[v] * inv);
    orow[32 + lm] = (bf16)(o2[v] * inv);
    orow[48 + lm] = (bf16)(o3[v] * inv);
  }
}

// ---------------------------------------------------------------------------
// Router: logits, softmax, top-2, counts (int atomics), probs for aux loss
// ---------------------------------------------------------------------------
__global__ void router_kernel(const bf16* __restrict__ Hn, const float* __restrict__ Wr,
                              float* __restrict__ probs, int* __restrict__ topi,
                              float* __restrict__ topp, int* __restrict__ counts) {
  const int t = blockIdx.x;
  const int tid = threadIdx.x;
  __shared__ float red[256 * En];
  float acc[En] = {};
  const bf16* xr = Hn + (size_t)t * Cn;
  for (int c = tid; c < Cn; c += 256) {
    const float xv = (float)xr[c];
    const float* wr = Wr + (size_t)c * En;
#pragma unroll
    for (int e = 0; e < En; ++e) acc[e] += xv * wr[e];
  }
#pragma unroll
  for (int e = 0; e < En; ++e) red[tid * En + e] = acc[e];
  __syncthreads();
  for (int st = 128; st > 0; st >>= 1) {
    if (tid < st)
#pragma unroll
      for (int e = 0; e < En; ++e) red[tid * En + e] += red[(tid + st) * En + e];
    __syncthreads();
  }
  if (tid == 0) {
    float lg[En]; float mx = -1e30f;
    for (int e = 0; e < En; ++e) { lg[e] = red[e]; mx = fmaxf(mx, lg[e]); }
    float sum = 0.f;
    for (int e = 0; e < En; ++e) { lg[e] = __expf(lg[e] - mx); sum += lg[e]; }
    const float inv = 1.f / sum;
    int i0 = 0;
    for (int e = 0; e < En; ++e) {
      lg[e] *= inv;
      probs[(size_t)t * En + e] = lg[e];
      if (lg[e] > lg[i0]) i0 = e;
    }
    int i1 = (i0 == 0) ? 1 : 0;
    for (int e = 0; e < En; ++e)
      if (e != i0 && lg[e] > lg[i1]) i1 = e;
    const float pn = 1.f / (lg[i0] + lg[i1]);
    topi[t * 2] = i0;          topi[t * 2 + 1] = i1;
    topp[t * 2] = lg[i0] * pn; topp[t * 2 + 1] = lg[i1] * pn;
    atomicAdd(&counts[i0], 1);
    atomicAdd(&counts[i1], 1);
  }
}

__global__ void init_counts(int* __restrict__ counts) {
  if (threadIdx.x < En) counts[threadIdx.x] = 0;
}

// Deterministic aux-loss reduction + expert offset scan + cursor reset
__global__ void aux_scan_kernel(const float* __restrict__ probs,
                                const int* __restrict__ topi,
                                const int* __restrict__ counts,
                                int* __restrict__ offs, int* __restrict__ cursors,
                                float* __restrict__ aux_out) {
  const int tid = threadIdx.x;
  __shared__ float fred[256 * En];
  __shared__ float pred[256 * En];
  float f[En] = {}, p[En] = {};
  for (int t = tid; t < Tn; t += 256) {
    const float* pr = probs + (size_t)t * En;
#pragma unroll
    for (int e = 0; e < En; ++e) p[e] += pr[e];
    f[topi[t * 2]] += 1.f;
  }
#pragma unroll
  for (int e = 0; e < En; ++e) { fred[tid * En + e] = f[e]; pred[tid * En + e] = p[e]; }
  __syncthreads();
  for (int st = 128; st > 0; st >>= 1) {
    if (tid < st)
#pragma unroll
      for (int e = 0; e < En; ++e) {
        fred[tid * En + e] += fred[(tid + st) * En + e];
        pred[tid * En + e] += pred[(tid + st) * En + e];
      }
    __syncthreads();
  }
  if (tid == 0) {
    int o = 0;
    for (int e = 0; e < En; ++e) { offs[e] = o; o += counts[e]; cursors[e] = 0; }
    offs[En] = o;
    float s = 0.f;
    for (int e = 0; e < En; ++e)
      s += (fred[e] / (float)Tn) * (pred[e] / (float)Tn);
    aux_out[0] = 0.01f * (float)En * s;
  }
}

__global__ void scatter_kernel(const int* __restrict__ topi, const float* __restrict__ topp,
                               const int* __restrict__ offs, int* __restrict__ cursors,
                               int* __restrict__ tok, float* __restrict__ gate,
                               int* __restrict__ slot_of) {
  const int t = blockIdx.x * 256 + threadIdx.x;
  if (t >= Tn) return;
  for (int j = 0; j < 2; ++j) {
    const int e = topi[t * 2 + j];
    const int pos = atomicAdd(&cursors[e], 1);
    const int slot = offs[e] + pos;
    tok[slot] = t;
    gate[slot] = topp[t * 2 + j];
    slot_of[t * 2 + j] = slot;
  }
}

// ---------------------------------------------------------------------------
// MoE up: hid = silu(X@w1[e]) * (X@w2[e]); fused dual-B GEMM, token gather,
// tile 128x64, bf16 out.
// ---------------------------------------------------------------------------
__global__ void moe_up_gemm(const bf16* __restrict__ X, const float* __restrict__ W1g,
                            const float* __restrict__ W2g, bf16* __restrict__ Hid,
                            const int* __restrict__ counts, const int* __restrict__ offs,
                            const int* __restrict__ tok) {
  const int e = blockIdx.z;
  const int cnt = counts[e];
  const int m0 = blockIdx.y * 128;
  if (m0 >= cnt) return;
  const int base = offs[e];
  const int n0 = blockIdx.x * 64;
  const float* B1 = W1g + (size_t)e * Cn * HIDn;
  const float* B2 = W2g + (size_t)e * Cn * HIDn;
  __shared__ __align__(16) bf16 sA[128 * 32];
  __shared__ __align__(16) bf16 sB1[64 * 32];
  __shared__ __align__(16) bf16 sB2[64 * 32];
  const int tid = threadIdx.x;
  const int lane = tid & 31, w = tid >> 5;
  const int lm = lane & 15, half = lane >> 4;
  const int ar = tid >> 1, ac = (tid & 1) * 16;
  const int bk = tid >> 3, bn = (tid & 7) * 8;
  int sl = m0 + ar; if (sl >= cnt) sl = cnt - 1;
  const bf16* arow = X + (size_t)tok[base + sl] * Cn;
  v8f zero = {};
  v8f a1[4] = {zero, zero, zero, zero};
  v8f a2[4] = {zero, zero, zero, zero};
  for (int kc = 0; kc < Cn; kc += 32) {
    uint4 av0 = *(const uint4*)(arow + kc + ac);
    uint4 av1 = *(const uint4*)(arow + kc + ac + 8);
    const float* p1 = B1 + (size_t)(kc + bk) * HIDn + n0 + bn;
    const float* p2 = B2 + (size_t)(kc + bk) * HIDn + n0 + bn;
    float4 x0 = *(const float4*)p1, x1 = *(const float4*)(p1 + 4);
    float4 y0 = *(const float4*)p2, y1 = *(const float4*)(p2 + 4);
    if (kc + 32 < Cn) {
      __builtin_prefetch(p1 + (size_t)32 * HIDn, 0, 1);
      __builtin_prefetch(p2 + (size_t)32 * HIDn, 0, 1);
    }
    __syncthreads();
    *(uint4*)(sA + ar * 32 + ac) = av0;
    *(uint4*)(sA + ar * 32 + ac + 8) = av1;
    bf16* q1 = sB1 + bk; bf16* q2 = sB2 + bk;
    q1[(bn + 0) * 32] = (bf16)x0.x; q1[(bn + 1) * 32] = (bf16)x0.y;
    q1[(bn + 2) * 32] = (bf16)x0.z; q1[(bn + 3) * 32] = (bf16)x0.w;
    q1[(bn + 4) * 32] = (bf16)x1.x; q1[(bn + 5) * 32] = (bf16)x1.y;
    q1[(bn + 6) * 32] = (bf16)x1.z; q1[(bn + 7) * 32] = (bf16)x1.w;
    q2[(bn + 0) * 32] = (bf16)y0.x; q2[(bn + 1) * 32] = (bf16)y0.y;
    q2[(bn + 2) * 32] = (bf16)y0.z; q2[(bn + 3) * 32] = (bf16)y0.w;
    q2[(bn + 4) * 32] = (bf16)y1.x; q2[(bn + 5) * 32] = (bf16)y1.y;
    q2[(bn + 6) * 32] = (bf16)y1.z; q2[(bn + 7) * 32] = (bf16)y1.w;
    __syncthreads();
    v16bf a = ld_frag(sA, w * 16 + lm, 32, 0, half);
#pragma unroll
    for (int j = 0; j < 4; ++j) {
      a1[j] = wmma_bf16(a, ld_frag(sB1, j * 16 + lm, 32, 0, half), a1[j]);
      a2[j] = wmma_bf16(a, ld_frag(sB2, j * 16 + lm, 32, 0, half), a2[j]);
    }
  }
#pragma unroll
  for (int j = 0; j < 4; ++j) {
#pragma unroll
    for (int v = 0; v < 8; ++v) {
      const int slot = m0 + w * 16 + v + 8 * half;
      if (slot < cnt) {
        const size_t o = (size_t)(base + slot) * HIDn + n0 + j * 16 + lm;
        const float h1 = a1[j][v], h2 = a2[j][v];
        Hid[o] = (bf16)((h1 / (1.f + __expf(-h1))) * h2);
      }
    }
  }
}

// ---------------------------------------------------------------------------
// MoE down: contrib[slot] = gate[slot] * (hid[slot] @ w3[e]); tile 128x64
// ---------------------------------------------------------------------------
__global__ void moe_down_gemm(const bf16* __restrict__ Hid, const float* __restrict__ W3g,
                              float* __restrict__ Contrib,
                              const int* __restrict__ counts, const int* __restrict__ offs,
                              const float* __restrict__ gate) {
  const int e = blockIdx.z;
  const int cnt = counts[e];
  const int m0 = blockIdx.y * 128;
  if (m0 >= cnt) return;
  const int base = offs[e];
  const int n0 = blockIdx.x * 64;
  const float* B = W3g + (size_t)e * HIDn * Cn;  // [HID, C]
  __shared__ __align__(16) bf16 sA[128 * 32];
  __shared__ __align__(16) bf16 sB[64 * 32];
  const int tid = threadIdx.x;
  const int lane = tid & 31, w = tid >> 5;
  const int lm = lane & 15, half = lane >> 4;
  const int ar = tid >> 1, ac = (tid & 1) * 16;
  const int bk = tid >> 3, bn = (tid & 7) * 8;
  int sl = m0 + ar; if (sl >= cnt) sl = cnt - 1;
  const bf16* arow = Hid + (size_t)(base + sl) * HIDn;
  v8f zero = {};
  v8f acc[4] = {zero, zero, zero, zero};
  for (int kc = 0; kc < HIDn; kc += 32) {
    uint4 av0 = *(const uint4*)(arow + kc + ac);
    uint4 av1 = *(const uint4*)(arow + kc + ac + 8);
    const float* bp = B + (size_t)(kc + bk) * Cn + n0 + bn;
    float4 b0 = *(const float4*)bp;
    float4 b1 = *(const float4*)(bp + 4);
    if (kc + 32 < HIDn) __builtin_prefetch(bp + (size_t)32 * Cn, 0, 1);
    __syncthreads();
    *(uint4*)(sA + ar * 32 + ac) = av0;
    *(uint4*)(sA + ar * 32 + ac + 8) = av1;
    bf16* q = sB + bk;
    q[(bn + 0) * 32] = (bf16)b0.x; q[(bn + 1) * 32] = (bf16)b0.y;
    q[(bn + 2) * 32] = (bf16)b0.z; q[(bn + 3) * 32] = (bf16)b0.w;
    q[(bn + 4) * 32] = (bf16)b1.x; q[(bn + 5) * 32] = (bf16)b1.y;
    q[(bn + 6) * 32] = (bf16)b1.z; q[(bn + 7) * 32] = (bf16)b1.w;
    __syncthreads();
    v16bf a = ld_frag(sA, w * 16 + lm, 32, 0, half);
#pragma unroll
    for (int j = 0; j < 4; ++j)
      acc[j] = wmma_bf16(a, ld_frag(sB, j * 16 + lm, 32, 0, half), acc[j]);
  }
#pragma unroll
  for (int j = 0; j < 4; ++j) {
#pragma unroll
    for (int v = 0; v < 8; ++v) {
      const int slot = m0 + w * 16 + v + 8 * half;
      if (slot < cnt) {
        const float g = gate[base + slot];
        Contrib[(size_t)(base + slot) * Cn + n0 + j * 16 + lm] = g * acc[j][v];
      }
    }
  }
}

// out[t] = h[t] + contrib[slot0(t)] + contrib[slot1(t)]
__global__ void combine_kernel(const float* __restrict__ H, const float* __restrict__ Contrib,
                               const int* __restrict__ slot_of, float* __restrict__ Out) {
  const int t = blockIdx.x;
  const int tid = threadIdx.x;
  const int s0 = slot_of[t * 2], s1 = slot_of[t * 2 + 1];
  for (int c = tid; c < Cn; c += 256)
    Out[(size_t)t * Cn + c] = H[(size_t)t * Cn + c] +
                              Contrib[(size_t)s0 * Cn + c] +
                              Contrib[(size_t)s1 * Cn + c];
}

// ---------------------------------------------------------------------------
extern "C" void kernel_launch(void* const* d_in, const int* in_sizes, int n_in,
                              void* d_out, int out_size, void* d_ws, size_t ws_size,
                              hipStream_t stream) {
  (void)in_sizes; (void)n_in; (void)out_size; (void)ws_size;
  const float* x        = (const float*)d_in[0];
  const float* attn_w   = (const float*)d_in[1];
  const float* wq       = (const float*)d_in[2];
  const float* wk       = (const float*)d_in[3];
  const float* wv       = (const float*)d_in[4];
  const float* wo       = (const float*)d_in[5];
  const float* mlp_w    = (const float*)d_in[6];
  const float* router_w = (const float*)d_in[7];
  const float* w1       = (const float*)d_in[8];
  const float* w2       = (const float*)d_in[9];
  const float* w3       = (const float*)d_in[10];

  char* ws = (char*)d_ws;
  size_t off = 0;
  auto alloc = [&](size_t bytes) -> void* {
    void* p = ws + off;
    off = (off + bytes + 255) & ~(size_t)255;
    return p;
  };

  bf16*  xn      = (bf16*)alloc((size_t)Tn * Cn * 2);
  float* qf      = (float*)alloc((size_t)Tn * Cn * 4);
  float* kf      = (float*)alloc((size_t)Tn * Cn * 4);
  float* vf      = (float*)alloc((size_t)Tn * Cn * 4);
  bf16*  qh      = (bf16*)alloc((size_t)Tn * Cn * 2);
  bf16*  kh      = (bf16*)alloc((size_t)Tn * Cn * 2);
  bf16*  vh      = (bf16*)alloc((size_t)Tn * Cn * 2);
  bf16*  att     = (bf16*)alloc((size_t)Tn * Cn * 2);
  float* h       = (float*)alloc((size_t)Tn * Cn * 4);
  bf16*  hn      = (bf16*)alloc((size_t)Tn * Cn * 2);
  bf16*  hid     = (bf16*)alloc((size_t)Tn * 2 * HIDn * 2);
  float* contrib = (float*)alloc((size_t)Tn * 2 * Cn * 4);
  float* probs   = (float*)alloc((size_t)Tn * En * 4);
  int*   topi    = (int*)alloc((size_t)Tn * 2 * 4);
  float* topp    = (float*)alloc((size_t)Tn * 2 * 4);
  int*   counts  = (int*)alloc(En * 4);
  int*   offs    = (int*)alloc((En + 1) * 4);
  int*   cursors = (int*)alloc(En * 4);
  int*   tok     = (int*)alloc((size_t)Tn * 2 * 4);
  float* gate    = (float*)alloc((size_t)Tn * 2 * 4);
  int*   slot_of = (int*)alloc((size_t)Tn * 2 * 4);

  float* out = (float*)d_out;
  float* aux = out + (size_t)Tn * Cn;

  const dim3 gemm_grid(Cn / 64, Tn / 128);

  init_counts<<<1, 32, 0, stream>>>(counts);
  rmsnorm_bf16<<<Tn, 256, 0, stream>>>(x, attn_w, xn);
  gemm_bf16_f32<<<gemm_grid, 256, 0, stream>>>(xn, wq, qf, nullptr, Cn, Cn);
  gemm_bf16_f32<<<gemm_grid, 256, 0, stream>>>(xn, wk, kf, nullptr, Cn, Cn);
  gemm_bf16_f32<<<gemm_grid, 256, 0, stream>>>(xn, wv, vf, nullptr, Cn, Cn);
  rope_pack<<<Tn, 512, 0, stream>>>(qf, kf, vf, qh, kh, vh);
  flash_attn<<<dim3(Tn / 64, NHn), 128, 0, stream>>>(qh, kh, vh, att);
  gemm_bf16_f32<<<gemm_grid, 256, 0, stream>>>(att, wo, h, x, Cn, Cn);
  rmsnorm_bf16<<<Tn, 256, 0, stream>>>(h, mlp_w, hn);
  router_kernel<<<Tn, 256, 0, stream>>>(hn, router_w, probs, topi, topp, counts);
  aux_scan_kernel<<<1, 256, 0, stream>>>(probs, topi, counts, offs, cursors, aux);
  scatter_kernel<<<Tn / 256, 256, 0, stream>>>(topi, topp, offs, cursors, tok, gate, slot_of);
  moe_up_gemm<<<dim3(HIDn / 64, 16, En), 256, 0, stream>>>(hn, w1, w2, hid, counts, offs, tok);
  moe_down_gemm<<<dim3(Cn / 64, 16, En), 256, 0, stream>>>(hid, w3, contrib, counts, offs, gate);
  combine_kernel<<<Tn, 256, 0, stream>>>(h, contrib, slot_of, out);
}
